// MAB_66194035966469
// MI455X (gfx1250) — 16-bit and 8-bit matrix operand order repaired and checked by reading, compiled, not run on hardware
//
#include <hip/hip_runtime.h>
#include <hip/hip_bf16.h>


typedef __attribute__((ext_vector_type(16))) _Float16 v16h;
typedef __attribute__((ext_vector_type(4)))  _Float16 v4h;
typedef __attribute__((ext_vector_type(8)))  float    v8f;
typedef __attribute__((ext_vector_type(8)))  int      v8i;

#define DMODEL 1024
#define NHEADS 16
#define HDIM   64
#define BATCH  8
#define SEQ    1024
#define ROWS   (BATCH * SEQ)

__device__ __forceinline__ void async_ld16(void* lds, const void* g) {
  asm volatile("global_load_async_to_lds_b128 %0, %1, off"
               :: "v"((unsigned)(unsigned long long)(uintptr_t)lds),
                  "v"((unsigned long long)(uintptr_t)g)
               : "memory");
}
__device__ __forceinline__ void wait_async4() { asm volatile("s_wait_asynccnt 0x4" ::: "memory"); }
__device__ __forceinline__ void wait_async2() { asm volatile("s_wait_asynccnt 0x2" ::: "memory"); }
__device__ __forceinline__ void wait_async0() { asm volatile("s_wait_asynccnt 0x0" ::: "memory"); }


__device__ inline v16h frag_a_h(const _Float16* __restrict__ base, int lane, int ld) {
  v16h a;
  const int m  = lane & 15;
  const int kb = (lane >> 4) * 16;
  const _Float16* row = base + (size_t)m * ld;
#pragma unroll
  for (int v = 0; v < 8; ++v) {
    const int k = (v << 1) + kb;
    a[2 * v]     = row[k];
    a[2 * v + 1] = row[k + 1];
  }
  return a;
}

__device__ inline v16h frag_b_h(const _Float16* __restrict__ base, int lane, int ld) {
  v16h b;
  const int n  = lane & 15;
  const int kb = (lane >> 4) * 16;
  const _Float16* row = base + (size_t)n * ld + kb;
#pragma unroll
  for (int h = 0; h < 16; ++h) b[h] = row[h];
  return b;
}

__device__ inline v16h frag_b_h_rowmajor(const _Float16* __restrict__ base, int lane, int ld) {
  v16h b;
  const int n  = lane & 15;
  const int kb = (lane >> 4) * 16;
#pragma unroll
  for (int h = 0; h < 16; ++h) b[h] = base[(size_t)(kb + h) * ld + n];
  return b;
}

__device__ inline v8i frag_a_u8(const unsigned char* __restrict__ base, int lane, int ld) {
  v8i a;
  const int m  = lane & 15;
  const int kb = (lane >> 4) * 8;
  const unsigned char* row = base + (size_t)m * ld;
#pragma unroll
  for (int v = 0; v < 8; ++v) {
    const int k = ((v & 1) << 2) + (((v >> 1) & 1) << 4) + ((v >> 2) << 5) + kb;
    a[v] = *(const int*)(row + k);
  }
  return a;
}

__device__ inline v8i frag_b_u8(const unsigned char* __restrict__ base, int lane, int ld) {
  v8i b;
  const int n  = lane & 15;
  const int kg = (lane >> 4) * 8;
  const unsigned char* row = base + (size_t)n * ld;
#pragma unroll
  for (int v = 0; v < 8; ++v) {
    const int k = ((v & 1) << 2) + (((v >> 1) & 1) << 4) + ((v >> 2) << 5) + kg;
    b[v] = *(const int*)(row + k);
  }
  return b;
}

__global__ __launch_bounds__(256) void lif_half4_kernel(const float* __restrict__ x,
                                                        _Float16* __restrict__ y) {
  const int i = (blockIdx.x * 256 + threadIdx.x) * 4;
  const float4 v = *(const float4*)(x + i);
  v4h o;
  o[0] = (_Float16)((v.x >= 2.0f) ? 1.0f : 0.0f);
  o[1] = (_Float16)((v.y >= 2.0f) ? 1.0f : 0.0f);
  o[2] = (_Float16)((v.z >= 2.0f) ? 1.0f : 0.0f);
  o[3] = (_Float16)((v.w >= 2.0f) ? 1.0f : 0.0f);
  *(v4h*)(y + i) = o;
}

__global__ __launch_bounds__(256) void f32_to_f16_kernel(const float* __restrict__ x,
                                                         _Float16* __restrict__ y) {
  const int i = (blockIdx.x * 256 + threadIdx.x) * 4;
  const float4 v = *(const float4*)(x + i);
  v4h o;
  o[0] = (_Float16)v.x; o[1] = (_Float16)v.y; o[2] = (_Float16)v.z; o[3] = (_Float16)v.w;
  *(v4h*)(y + i) = o;
}

#define LDT 40

template <int MODE>
__global__ __launch_bounds__(256) void gemm_kernel(const _Float16* __restrict__ A,
                                                   const _Float16* __restrict__ Wh,
                                                   const float* __restrict__ bias,
                                                   const float* __restrict__ skip,
                                                   void* __restrict__ out) {
  __shared__ _Float16 Ah[2][128][LDT];
  __shared__ _Float16 Bh[2][128][LDT];

  const int tid  = threadIdx.x;
  const int lane = tid & 31;
  const int wid  = tid >> 5;
  const int wm   = (wid & 3) * 32;
  const int wn   = (wid >> 2) * 64;
  const int m0   = blockIdx.y * 128;
  const int n0   = blockIdx.x * 128;

  const int c0    = tid * 2;
  const int arow0 = c0 >> 2,       akoff0 = (c0 & 3) * 8;
  const int arow1 = (c0 + 1) >> 2, akoff1 = ((c0 + 1) & 3) * 8;

  auto stage = [&](int buf, int k0) {
    async_ld16(&Ah[buf][arow0][akoff0], A  + (size_t)(m0 + arow0) * DMODEL + k0 + akoff0);
    async_ld16(&Ah[buf][arow1][akoff1], A  + (size_t)(m0 + arow1) * DMODEL + k0 + akoff1);
    async_ld16(&Bh[buf][arow0][akoff0], Wh + (size_t)(n0 + arow0) * DMODEL + k0 + akoff0);
    async_ld16(&Bh[buf][arow1][akoff1], Wh + (size_t)(n0 + arow1) * DMODEL + k0 + akoff1);
  };

  v8f acc[2][4] = {};

  stage(0, 0);
  for (int it = 0; it < DMODEL / 32; ++it) {
    const int buf = it & 1;
    if (it + 1 < DMODEL / 32) {
      stage(buf ^ 1, (it + 1) * 32);
      wait_async4();
    } else {
      wait_async0();
    }
    __syncthreads();

    const v16h a0 = frag_a_h(&Ah[buf][wm][0],      lane, LDT);
    const v16h a1 = frag_a_h(&Ah[buf][wm + 16][0], lane, LDT);
#pragma unroll
    for (int j = 0; j < 4; ++j) {
      const v16h bj = frag_b_h(&Bh[buf][wn + 16 * j][0], lane, LDT);
      acc[0][j] = __builtin_amdgcn_wmma_f32_16x16x32_f16(false, a0, false, bj, (short)0, acc[0][j], false, false);
      acc[1][j] = __builtin_amdgcn_wmma_f32_16x16x32_f16(false, a1, false, bj, (short)0, acc[1][j], false, false);
    }
    __syncthreads();
  }

  const int crow = (lane >> 4) << 3;
  const int ccol = lane & 15;
#pragma unroll
  for (int i = 0; i < 2; ++i) {
#pragma unroll
    for (int j = 0; j < 4; ++j) {
      const int gn = n0 + wn + j * 16 + ccol;
      const float bn = bias[gn];
#pragma unroll
      for (int r = 0; r < 8; ++r) {
        const int gm = m0 + wm + i * 16 + crow + r;
        const size_t idx = (size_t)gm * DMODEL + gn;
        const float v = acc[i][j][r] + bn;
        if constexpr (MODE == 0) {
          ((float*)out)[idx] = v;
        } else if constexpr (MODE == 1) {
          ((_Float16*)out)[idx] = (_Float16)v;
        } else {
          const float sp = (v > 15.0f) ? v : log1pf(__expf(v));
          ((float*)out)[idx] = skip[idx] + v * tanhf(sp);
        }
      }
    }
  }
}

__global__ __launch_bounds__(256) void ln_lif_kernel(const float* __restrict__ x,
                                                     const float* __restrict__ g,
                                                     const float* __restrict__ be,
                                                     unsigned char* __restrict__ y) {
  __shared__ float rs[256];
  __shared__ float rq[256];
  const int tid = threadIdx.x;
  const size_t row = blockIdx.x;
  const float* xr = x + row * DMODEL;

  float vals[4];
  float s = 0.f, s2 = 0.f;
#pragma unroll
  for (int i = 0; i < 4; ++i) {
    const float v = xr[tid + i * 256];
    vals[i] = v;
    s += v;
    s2 += v * v;
  }
  rs[tid] = s;
  rq[tid] = s2;
  __syncthreads();
  for (int off = 128; off > 0; off >>= 1) {
    if (tid < off) { rs[tid] += rs[tid + off]; rq[tid] += rq[tid + off]; }
    __syncthreads();
  }
  const float mu  = rs[0] * (1.0f / DMODEL);
  const float var = rq[0] * (1.0f / DMODEL) - mu * mu;
  const float inv = rsqrtf(var + 1e-5f);
#pragma unroll
  for (int i = 0; i < 4; ++i) {
    const int col = tid + i * 256;
    const float yv = (vals[i] - mu) * inv * g[col] + be[col];
    y[row * DMODEL + col] = (yv >= 2.0f) ? 1 : 0;
  }
}

#define S_LD 1024
#define ATTN_VT_OFF (16 * 1024 * 6 + (128 + 16 + 16) * 4)
#define ATTN_LDS    (ATTN_VT_OFF + 2 * 32 * 64 * 2)

__global__ __launch_bounds__(128) void attn_kernel(const unsigned char* __restrict__ q8,
                                                   const unsigned char* __restrict__ k8,
                                                   const _Float16* __restrict__ vh,
                                                   const float* __restrict__ mask,
                                                   float* __restrict__ outf,
                                                   _Float16* __restrict__ outh) {
  extern __shared__ char smem[];
  float*    s    = (float*)smem;
  _Float16* p    = (_Float16*)(smem + 16 * 1024 * 4);
  float*    red  = (float*)(smem + 16 * 1024 * 6);
  float*    rowm = red + 128;
  float*    rowl = rowm + 16;
  _Float16* Vt   = (_Float16*)(smem + ATTN_VT_OFF);

  const int qt   = blockIdx.x;
  const int h    = blockIdx.y;
  const int b    = blockIdx.z;
  const int tid  = threadIdx.x;
  const int wid  = tid >> 5;
  const int lane = tid & 31;
  const int q0   = qt * 16;

  const unsigned char* qbase = q8 + ((size_t)b * SEQ + q0) * DMODEL + h * HDIM;
  const unsigned char* kbase = k8 + (size_t)b * SEQ * DMODEL + h * HDIM;
  const float*         mbase = mask + ((size_t)b * SEQ + q0) * SEQ;
  const _Float16*      vhead = vh + (size_t)b * SEQ * DMODEL + h * HDIM;

  const int crow = (lane >> 4) << 3;
  const int ccol = lane & 15;

  auto stageV = [&](int buf, int kv0) {
#pragma unroll
    for (int cc = 0; cc < 2; ++cc) {
      const int ch = tid * 2 + cc;
      const int kr = ch >> 3;
      const int co = (ch & 7) * 8;
      async_ld16(Vt + ((size_t)buf * 32 + kr) * 64 + co,
                 vhead + (size_t)(kv0 + kr) * DMODEL + co);
    }
  };
  stageV(0, 0);

  const v8i afrag = frag_a_u8(qbase, lane, DMODEL);

  for (int t = wid; t < SEQ / 16; t += 4) {
    const int kv0 = t * 16;
    const v8i bfrag = frag_b_u8(kbase + (size_t)kv0 * DMODEL, lane, DMODEL);
    v8i c = {};
    c = __builtin_amdgcn_wmma_i32_16x16x64_iu8(false, afrag, false, bfrag, c, false, false);
#pragma unroll
    for (int r = 0; r < 8; ++r) {
      const int row = crow + r;
      const float sv = (float)c[r] * 0.125f * mbase[(size_t)row * SEQ + kv0 + ccol];
      s[row * S_LD + kv0 + ccol] = sv;
    }
  }
  __syncthreads();

  {
    const int row = tid >> 3;
    const int sub = tid & 7;
    const float* sr = s + row * S_LD + sub * 128;
    float mx = -1e30f;
    for (int i = 0; i < 128; ++i) mx = fmaxf(mx, sr[i]);
    red[row * 8 + sub] = mx;
    __syncthreads();
    if (sub == 0) {
      float m2 = red[row * 8];
#pragma unroll
      for (int i = 1; i < 8; ++i) m2 = fmaxf(m2, red[row * 8 + i]);
      rowm[row] = m2;
    }
    __syncthreads();
    const float m2 = rowm[row];
    _Float16* pr = p + row * S_LD + sub * 128;
    float lsum = 0.f;
    for (int i = 0; i < 128; ++i) {
      const float e = __expf(sr[i] - m2);
      lsum += e;
      pr[i] = (_Float16)e;
    }
    __syncthreads();
    red[row * 8 + sub] = lsum;
    __syncthreads();
    if (sub == 0) {
      float l2 = 0.f;
#pragma unroll
      for (int i = 0; i < 8; ++i) l2 += red[row * 8 + i];
      rowl[row] = l2;
    }
  }
  __syncthreads();

  const int n0 = wid * 16;
  v8f acc = {};
  for (int it = 0; it < SEQ / 32; ++it) {
    const int buf = it & 1;
    if (it + 1 < SEQ / 32) {
      stageV(buf ^ 1, (it + 1) * 32);
      wait_async2();
    } else {
      wait_async0();
    }
    __syncthreads();
    const v16h pa = frag_a_h(p + it * 32, lane, S_LD);
    const v16h vb = frag_b_h_rowmajor(Vt + (size_t)buf * 32 * 64 + n0, lane, 64);
    acc = __builtin_amdgcn_wmma_f32_16x16x32_f16(false, pa, false, vb, (short)0, acc, false, false);
    __syncthreads();
  }
#pragma unroll
  for (int r = 0; r < 8; ++r) {
    const int row = crow + r;
    const float val = acc[r] / rowl[row];
    const size_t idx = ((size_t)b * SEQ + q0 + row) * DMODEL + h * HDIM + n0 + ccol;
    outf[idx] = val;
    outh[idx] = (_Float16)val;
  }
}

extern "C" void kernel_launch(void* const* d_in, const int* in_sizes, int n_in,
                              void* d_out, int out_size, void* d_ws, size_t ws_size,
                              hipStream_t stream) {
  const float* Q    = (const float*)d_in[0];
  const float* K    = (const float*)d_in[1];
  const float* mask = (const float*)d_in[2];
  const float* Wq   = (const float*)d_in[3];
  const float* bq   = (const float*)d_in[4];
  const float* Wk   = (const float*)d_in[5];
  const float* bk   = (const float*)d_in[6];
  const float* Wv   = (const float*)d_in[7];
  const float* bv   = (const float*)d_in[8];
  const float* Wo   = (const float*)d_in[9];
  const float* bo   = (const float*)d_in[10];
  const float* gq   = (const float*)d_in[11];
  const float* beq  = (const float*)d_in[12];
  const float* gk   = (const float*)d_in[13];
  const float* bek  = (const float*)d_in[14];

  const size_t NE = (size_t)ROWS * DMODEL;
  const size_t NW = (size_t)DMODEL * DMODEL;
  char* ws = (char*)d_ws;
  _Float16*      qh   = (_Float16*)ws;       ws += NE * 2;
  _Float16*      kh   = (_Float16*)ws;       ws += NE * 2;
  float*         pre  = (float*)ws;          ws += NE * 4;
  _Float16*      vh   = (_Float16*)ws;       ws += NE * 2;
  unsigned char* q8   = (unsigned char*)ws;  ws += NE;
  unsigned char* k8   = (unsigned char*)ws;  ws += NE;
  float*         outf = (float*)ws;          ws += NE * 4;
  _Float16*      outh = (_Float16*)ws;       ws += NE * 2;
  _Float16*      Wqh  = (_Float16*)ws;       ws += NW * 2;
  _Float16*      Wkh  = (_Float16*)ws;       ws += NW * 2;
  _Float16*      Wvh  = (_Float16*)ws;       ws += NW * 2;
  _Float16*      Woh  = (_Float16*)ws;       ws += NW * 2;

  const dim3 gemm_grid(DMODEL / 128, ROWS / 128);
  const int  wconv_blocks = (int)(NW / (256 * 4));

  lif_half4_kernel<<<NE / (256 * 4), 256, 0, stream>>>(Q, qh);
  lif_half4_kernel<<<NE / (256 * 4), 256, 0, stream>>>(K, kh);
  f32_to_f16_kernel<<<wconv_blocks, 256, 0, stream>>>(Wq, Wqh);
  f32_to_f16_kernel<<<wconv_blocks, 256, 0, stream>>>(Wk, Wkh);
  f32_to_f16_kernel<<<wconv_blocks, 256, 0, stream>>>(Wv, Wvh);
  f32_to_f16_kernel<<<wconv_blocks, 256, 0, stream>>>(Wo, Woh);

  gemm_kernel<0><<<gemm_grid, 256, 0, stream>>>(qh, Wqh, bq, nullptr, pre);
  ln_lif_kernel<<<ROWS, 256, 0, stream>>>(pre, gq, beq, q8);

  gemm_kernel<0><<<gemm_grid, 256, 0, stream>>>(kh, Wkh, bk, nullptr, pre);
  ln_lif_kernel<<<ROWS, 256, 0, stream>>>(pre, gk, bek, k8);

  gemm_kernel<1><<<gemm_grid, 256, 0, stream>>>(kh, Wvh, bv, nullptr, vh);

  attn_kernel<<<dim3(SEQ / 16, NHEADS, BATCH), 128, ATTN_LDS, stream>>>(
      q8, k8, vh, mask, outf, outh);

  gemm_kernel<2><<<gemm_grid, 256, 0, stream>>>(outh, Woh, bo, outf, (float*)d_out);

  (void)in_sizes; (void)n_in; (void)out_size; (void)ws_size;
}
